// DMPNN_73504070304139
// MI455X (gfx1250) — compile-verified, matrix operand order checked by reading, not run on hardware
//
#include <hip/hip_runtime.h>
#include <hip/hip_bf16.h>

// ---------------------------------------------------------------------------
// DMPNN (NNConv + GRU message passing) for MI455X / gfx1250, wave32.
//
// Roofline: Wm[E,32,32] dominates bytes (bf16: 307 MB, 1 write + 3 reads
// ~1.23 GB -> ~53us @ 23.3 TB/s). Edge-net layer2 dominates FLOPs
// (E x 128 x 1024 GEMM = 39 GFLOP) -> v_wmma_f32_16x16x32_bf16.
// Everything else is small VALU work fused per-wave.
// ---------------------------------------------------------------------------

#define NNODES 50000
#define NEDGES 150000
#define NODE_IN 64
#define EDGE_IN 16
#define DD 32
#define EH 128
#define DD2 1024   // D*D
#define STEPS 3

typedef __attribute__((ext_vector_type(16))) __bf16 v16bf;
typedef __attribute__((ext_vector_type(8)))  float  v8f;

// ---------------------------------------------------------------------------
// Prep: We2 [128][1024] f32 (k-major) -> We2T [1024][128] bf16 (n-major).
// Makes B-fragment loads two contiguous 16B loads per lane.
// ---------------------------------------------------------------------------
__global__ void k_prep_we2t(const float* __restrict__ We2,
                            __bf16* __restrict__ We2T) {
    int idx = blockIdx.x * blockDim.x + threadIdx.x;  // over 128*1024
    if (idx >= EH * DD2) return;
    int k = idx / DD2;
    int c = idx % DD2;
    We2T[(size_t)c * EH + k] = (__bf16)We2[idx];
}

// ---------------------------------------------------------------------------
// Edge network layer 1: U[e,:] = relu(edge_attr[e,:] @ We1 + be1), bf16 out.
// One wave per edge; lane computes 4 of the 128 hidden units.
// ---------------------------------------------------------------------------
__global__ void k_edge_l1(const float* __restrict__ ea,
                          const float* __restrict__ We1,
                          const float* __restrict__ be1,
                          __bf16* __restrict__ U) {
    __shared__ float sW[EDGE_IN * EH];
    __shared__ float sb[EH];
    for (int t = threadIdx.x; t < EDGE_IN * EH; t += blockDim.x) sW[t] = We1[t];
    for (int t = threadIdx.x; t < EH; t += blockDim.x) sb[t] = be1[t];
    __syncthreads();

    int wave = threadIdx.x >> 5, lane = threadIdx.x & 31;
    int e = blockIdx.x * 8 + wave;
    if (e >= NEDGES) return;

    float av = (lane < EDGE_IN) ? ea[(size_t)e * EDGE_IN + lane] : 0.0f;
    float a0 = sb[lane], a1 = sb[32 + lane], a2 = sb[64 + lane], a3 = sb[96 + lane];
#pragma unroll
    for (int k = 0; k < EDGE_IN; ++k) {
        float a = __shfl(av, k, 32);
        a0 += a * sW[k * EH + lane];
        a1 += a * sW[k * EH + 32 + lane];
        a2 += a * sW[k * EH + 64 + lane];
        a3 += a * sW[k * EH + 96 + lane];
    }
    size_t base = (size_t)e * EH;
    U[base + lane]      = (__bf16)fmaxf(a0, 0.0f);
    U[base + 32 + lane] = (__bf16)fmaxf(a1, 0.0f);
    U[base + 64 + lane] = (__bf16)fmaxf(a2, 0.0f);
    U[base + 96 + lane] = (__bf16)fmaxf(a3, 0.0f);
}

// ---------------------------------------------------------------------------
// Edge network layer 2 via WMMA bf16: Wm[E,1024] = U[E,128] @ We2 + be2.
// Block = 256 threads (8 waves) = 16 edges (M) x 1024 cols (N).
// Each wave: 8 N-tiles of 16; K=128 -> 4x v_wmma_f32_16x16x32_bf16.
// A fragment (16x32 bf16), lane L: m=L%16, hi=L/16,
//   elems 0..7 -> K = kk + hi*8 + j ; elems 8..15 -> K = kk + 16 + hi*8 + (j-8)
// B fragment mirrors with N in place of M; We2T is n-major so both are
// two contiguous 16-byte loads per lane. No divergence (EXEC all ones).
// ---------------------------------------------------------------------------
__global__ void k_edge_l2_wmma(const __bf16* __restrict__ U,
                               const __bf16* __restrict__ We2T,
                               const float* __restrict__ be2,
                               __bf16* __restrict__ Wm) {
    __shared__ unsigned short sA[16 * EH];  // 16 edges x 128 K (bf16 bits) = 4KB

    int m0 = blockIdx.x * 16;
    // Cooperative stage of A: 2048 ushorts = 256 uint4, one per thread.
    ((uint4*)sA)[threadIdx.x] =
        ((const uint4*)(U + (size_t)m0 * EH))[threadIdx.x];
    __syncthreads();

    int wave = threadIdx.x >> 5, lane = threadIdx.x & 31;
    int ml = lane & 15, hi = lane >> 4;

#pragma unroll
    for (int nt = 0; nt < 8; ++nt) {
        int n0 = wave * 128 + nt * 16;
        int nc = n0 + ml;                     // this lane's N column
        v8f acc = {};
#pragma unroll
        for (int kk = 0; kk < EH; kk += 32) {
            union { v16bf v; uint4 q[2]; } a, b;
            const unsigned short* ap = &sA[ml * EH + kk + hi * 8];
            a.q[0] = *(const uint4*)ap;
            a.q[1] = *(const uint4*)(ap + 16);
            const __bf16* bp = We2T + (size_t)nc * EH + kk + hi * 8;
            b.q[0] = *(const uint4*)bp;
            b.q[1] = *(const uint4*)(bp + 16);
            acc = __builtin_amdgcn_wmma_f32_16x16x32_bf16(
                false, a.v, false, b.v, (short)0, acc, false, false);
        }
        float bias = be2[nc];
#pragma unroll
        for (int v = 0; v < 8; ++v) {
            int m = v + hi * 8;               // C layout: VGPR v -> row v (+8 for hi half)
            Wm[(size_t)(m0 + m) * DD2 + nc] = (__bf16)(acc[v] + bias);
        }
    }
}

// ---------------------------------------------------------------------------
// Node init: h = relu(x @ Wn + bn). One wave per node, lane = output feature.
// ---------------------------------------------------------------------------
__global__ void k_node_init(const float* __restrict__ x,
                            const float* __restrict__ Wn,
                            const float* __restrict__ bn,
                            float* __restrict__ h) {
    __shared__ float sW[NODE_IN * DD];
    __shared__ float sb[DD];
    for (int t = threadIdx.x; t < NODE_IN * DD; t += blockDim.x) sW[t] = Wn[t];
    for (int t = threadIdx.x; t < DD; t += blockDim.x) sb[t] = bn[t];
    __syncthreads();

    int wave = threadIdx.x >> 5, lane = threadIdx.x & 31;
    int n = blockIdx.x * 8 + wave;
    if (n >= NNODES) return;

    float x0 = x[(size_t)n * NODE_IN + lane];
    float x1 = x[(size_t)n * NODE_IN + 32 + lane];
    float acc = sb[lane];
#pragma unroll
    for (int i = 0; i < 32; ++i) {
        acc += __shfl(x0, i, 32) * sW[i * DD + lane];
        acc += __shfl(x1, i, 32) * sW[(32 + i) * DD + lane];
    }
    h[(size_t)n * DD + lane] = fmaxf(acc, 0.0f);
}

__global__ void k_zero(float* __restrict__ p, int n) {
    int i = blockIdx.x * blockDim.x + threadIdx.x;
    if (i < n) p[i] = 0.0f;
}

// ---------------------------------------------------------------------------
// Message + scatter-add: one wave per edge, lane = output feature.
// m[o] = sum_i h[src,i] * Wm[e,i,o];  atomicAdd into agg[dst,o].
// Wm row reads: 32 lanes x 2B consecutive = one 64B transaction per i.
// ---------------------------------------------------------------------------
__global__ void k_message(const int* __restrict__ ei,
                          const float* __restrict__ h,
                          const __bf16* __restrict__ Wm,
                          float* __restrict__ agg) {
    int wave = threadIdx.x >> 5, lane = threadIdx.x & 31;
    int e = blockIdx.x * 8 + wave;
    if (e >= NEDGES) return;

    int s = ei[e];
    int d = ei[NEDGES + e];
    float hv = h[(size_t)s * DD + lane];
    const __bf16* w = Wm + (size_t)e * DD2 + lane;
    float m = 0.0f;
#pragma unroll
    for (int i = 0; i < DD; ++i)
        m += __shfl(hv, i, 32) * (float)w[i * DD];
    atomicAdd(&agg[(size_t)d * DD + lane], m);
}

// ---------------------------------------------------------------------------
// Fused node update: conv = relu(agg + h@Wroot + bconv); h = GRU(conv, h).
// One wave per node, lane = feature index; weights staged in LDS (~29 KB).
// ---------------------------------------------------------------------------
__global__ void k_update(const float* __restrict__ agg,
                         const float* __restrict__ Wroot,
                         const float* __restrict__ bconv,
                         const float* __restrict__ Wih,
                         const float* __restrict__ Whh,
                         const float* __restrict__ bih,
                         const float* __restrict__ bhh,
                         float* __restrict__ h) {
    __shared__ float sWr[DD * DD];
    __shared__ float sWi[3 * DD * DD];
    __shared__ float sWh[3 * DD * DD];
    __shared__ float sbc[DD];
    __shared__ float sbi[3 * DD];
    __shared__ float sbh[3 * DD];
    for (int t = threadIdx.x; t < DD * DD; t += blockDim.x) sWr[t] = Wroot[t];
    for (int t = threadIdx.x; t < 3 * DD * DD; t += blockDim.x) { sWi[t] = Wih[t]; sWh[t] = Whh[t]; }
    for (int t = threadIdx.x; t < DD; t += blockDim.x) sbc[t] = bconv[t];
    for (int t = threadIdx.x; t < 3 * DD; t += blockDim.x) { sbi[t] = bih[t]; sbh[t] = bhh[t]; }
    __syncthreads();

    int wave = threadIdx.x >> 5, lane = threadIdx.x & 31;
    int n = blockIdx.x * 8 + wave;
    if (n >= NNODES) return;

    float hv = h[(size_t)n * DD + lane];
    float av = agg[(size_t)n * DD + lane];

    float root = 0.0f;
#pragma unroll
    for (int i = 0; i < DD; ++i)
        root += __shfl(hv, i, 32) * sWr[i * DD + lane];
    float conv = fmaxf(av + root + sbc[lane], 0.0f);

    // gi = conv @ Wih.T + bih ; gh = h @ Whh.T + bhh  (gate order r,z,n)
    float gir = sbi[lane], giz = sbi[DD + lane], gin = sbi[2 * DD + lane];
    float ghr = sbh[lane], ghz = sbh[DD + lane], ghn = sbh[2 * DD + lane];
#pragma unroll
    for (int o = 0; o < DD; ++o) {
        float c  = __shfl(conv, o, 32);
        float hh = __shfl(hv, o, 32);
        gir += c  * sWi[lane * DD + o];
        giz += c  * sWi[(DD + lane) * DD + o];
        gin += c  * sWi[(2 * DD + lane) * DD + o];
        ghr += hh * sWh[lane * DD + o];
        ghz += hh * sWh[(DD + lane) * DD + o];
        ghn += hh * sWh[(2 * DD + lane) * DD + o];
    }
    float r  = 1.0f / (1.0f + __expf(-(gir + ghr)));
    float z  = 1.0f / (1.0f + __expf(-(giz + ghz)));
    float nn = tanhf(gin + r * ghn);
    h[(size_t)n * DD + lane] = (1.0f - z) * nn + z * hv;
}

// ---------------------------------------------------------------------------
extern "C" void kernel_launch(void* const* d_in, const int* in_sizes, int n_in,
                              void* d_out, int out_size, void* d_ws, size_t ws_size,
                              hipStream_t stream) {
    const float* x     = (const float*)d_in[0];
    const int*   ei    = (const int*)  d_in[1];
    const float* ea    = (const float*)d_in[2];
    const float* Wn    = (const float*)d_in[3];
    const float* bn    = (const float*)d_in[4];
    const float* We1   = (const float*)d_in[5];
    const float* be1   = (const float*)d_in[6];
    const float* We2   = (const float*)d_in[7];
    const float* be2   = (const float*)d_in[8];
    const float* Wroot = (const float*)d_in[9];
    const float* bconv = (const float*)d_in[10];
    const float* Wih   = (const float*)d_in[11];
    const float* Whh   = (const float*)d_in[12];
    const float* bih   = (const float*)d_in[13];
    const float* bhh   = (const float*)d_in[14];

    // Workspace layout (all offsets 256B-aligned by construction).
    char* ws = (char*)d_ws;
    size_t off = 0;
    __bf16* Wm   = (__bf16*)(ws + off); off += (size_t)NEDGES * DD2 * 2;   // 307.2 MB
    __bf16* U    = (__bf16*)(ws + off); off += (size_t)NEDGES * EH * 2;    //  38.4 MB
    __bf16* We2T = (__bf16*)(ws + off); off += (size_t)EH * DD2 * 2;       //  0.25 MB
    float*  hbuf = (float*) (ws + off); off += (size_t)NNODES * DD * 4;    //   6.4 MB
    float*  agg  = (float*) (ws + off); off += (size_t)NNODES * DD * 4;    //   6.4 MB

    // 1) We2 -> bf16, n-major
    k_prep_we2t<<<(EH * DD2 + 255) / 256, 256, 0, stream>>>(We2, We2T);
    // 2) edge network layer 1
    k_edge_l1<<<(NEDGES + 7) / 8, 256, 0, stream>>>(ea, We1, be1, U);
    // 3) edge network layer 2: WMMA GEMM -> per-edge weight matrices (bf16)
    k_edge_l2_wmma<<<NEDGES / 16, 256, 0, stream>>>(U, We2T, be2, Wm);
    // 4) node init
    k_node_init<<<(NNODES + 7) / 8, 256, 0, stream>>>(x, Wn, bn, hbuf);

    // 5) message passing steps
    for (int s = 0; s < STEPS; ++s) {
        k_zero<<<(NNODES * DD + 255) / 256, 256, 0, stream>>>(agg, NNODES * DD);
        k_message<<<(NEDGES + 7) / 8, 256, 0, stream>>>(ei, hbuf, Wm, agg);
        k_update<<<(NNODES + 7) / 8, 256, 0, stream>>>(agg, Wroot, bconv,
                                                       Wih, Whh, bih, bhh, hbuf);
    }

    // 6) output
    hipMemcpyAsync(d_out, hbuf, (size_t)NNODES * DD * sizeof(float),
                   hipMemcpyDeviceToDevice, stream);
}
